// MIM_AE_34900904247922
// MI455X (gfx1250) — matrix operand order repaired and checked by reading, not compiled, not run on hardware
//
#include <hip/hip_runtime.h>

#define T_STEPS 6
#define D_IN 16

typedef __attribute__((ext_vector_type(16))) _Float16 v16h;
typedef __attribute__((ext_vector_type(8)))  _Float16 v8h;
typedef __attribute__((ext_vector_type(8)))  float    v8f;

union H16 { v16h v; v8h h[2]; };

// ---------------------------------------------------------------------------
// helpers
// ---------------------------------------------------------------------------
__device__ __forceinline__ v8f wmma_(v16h a, v16h b, v8f c) {
  return __builtin_amdgcn_wmma_f32_16x16x32_f16(
      /*neg_a=*/false, a, /*neg_b=*/false, b,
      /*c_mod=*/(short)0, c, /*reuse_a=*/false, /*reuse_b=*/false);
}

// A fragment (16xK tile, f16 row-major in LDS, row stride lda halves).
// Lane layout (ISA 7.12.2, 16-bit A 16x32): lanes 0-15: halves 0..7 = K0..7,
// halves 8..15 = K16..23; lanes 16-31: K8..15 / K24..31.
__device__ __forceinline__ v16h lda_frag(const _Float16* a, int lda, int kt,
                                         int m, int hi) {
  const _Float16* p = a + m * lda + kt * 32;
  H16 u;
  u.h[0] = *(const v8h*)(p + hi * 8);
  u.h[1] = *(const v8h*)(p + 16 + hi * 8);
  return u.v;
}

// B fragment (Kx16 weight stored transposed as [N][Kp] f16 in global).
// Lanes 0-15 (col n): halves 0..7 = K0..7, 8..15 = K16..23; lanes 16-31: K8..15 / K24..31.
__device__ __forceinline__ v16h ldb_frag(const _Float16* wt, int Kp, int ncol,
                                         int kt, int hi) {
  const _Float16* p = wt + (size_t)ncol * Kp + kt * 32 + hi * 8;
  H16 u;
  u.h[0] = *(const v8h*)(p);
  u.h[1] = *(const v8h*)(p + 16);
  return u.v;
}

__device__ __forceinline__ v8f vsplat(float v) {
  v8f r;
#pragma unroll
  for (int i = 0; i < 8; ++i) r[i] = v;
  return r;
}

// Hardware-transcendental activations: v_exp_f32 (exp2) + v_rcp_f32.
// Saturation is handled naturally: exp2(+inf)->inf, rcp(inf)->0.
#define LOG2E 1.4426950408889634f
__device__ __forceinline__ float sigf(float x) {
  return __builtin_amdgcn_rcpf(1.f + __builtin_amdgcn_exp2f(-x * LOG2E));
}
__device__ __forceinline__ float tanhf_(float x) {
  // tanh(x) = 1 - 2/(exp(2x)+1)
  return __builtin_fmaf(
      -2.f,
      __builtin_amdgcn_rcpf(1.f + __builtin_amdgcn_exp2f(x * (2.f * LOG2E))),
      1.f);
}
__device__ __forceinline__ v8f vsig(v8f x) {
  v8f r;
#pragma unroll
  for (int i = 0; i < 8; ++i) r[i] = sigf(x[i]);
  return r;
}
__device__ __forceinline__ v8f vtanh(v8f x) {
  v8f r;
#pragma unroll
  for (int i = 0; i < 8; ++i) r[i] = tanhf_(x[i]);
  return r;
}

// Store a 16x16 f32 D tile (N across lanes, M across vgprs) to LDS as f16.
__device__ __forceinline__ void st_tile(_Float16* a, int lda, int jt, int n,
                                        int mb, v8f d) {
#pragma unroll
  for (int r = 0; r < 8; ++r)
    a[(mb + r) * lda + jt * 16 + n] = (_Float16)d[r];
}

// ---------------------------------------------------------------------------
// weight preprocessing: f32 [K][N]  ->  f16 transposed padded [N][Kp] in ws
// ---------------------------------------------------------------------------
#define NMAT 31
struct PrepArgs {
  const float* src[NMAT];
  int K[NMAT];
  int N[NMAT];
  int off[NMAT];
};

__global__ __launch_bounds__(256) void MIM_prep_kernel(PrepArgs P,
                                                       _Float16* dst) {
  int id = blockIdx.x;
  const float* s = P.src[id];
  int K = P.K[id], N = P.N[id];
  int Kp = (K + 31) & ~31;
  _Float16* d = dst + P.off[id];
  int tot = N * Kp;
  for (int i = threadIdx.x; i < tot; i += 256) {
    int nn = i / Kp, kk = i - nn * Kp;
    float v = (kk < K) ? s[(size_t)kk * N + nn] : 0.f;
    d[i] = (_Float16)v;
  }
}

// ---------------------------------------------------------------------------
// main kernel
// ---------------------------------------------------------------------------
struct MainArgs {
  const float* x;
  float* out;
  const _Float16* wb;
  const float* b1[4];  // f,i,o,c  (H1=32)
  const float* b2[4];  // f1,i1,o1,c1 (HN=64)
  const float* b3[4];  // f2,i2,o2,c2 (HS=128)
  const float* b4[3];  // i3,o3,c3 (H2=128)
};

// LDS strides (halves), padded to stagger banks
#define L32 40
#define L64 72
#define L128 136
// per-wave LDS halves: 3*16*L32 + 2*16*L64 + 5*16*L128
#define PW (3 * 16 * L32 + 2 * 16 * L64 + 5 * 16 * L128)  // 15104

__global__ __launch_bounds__(64) void MIM_lstm_kernel(MainArgs A) {
  __shared__ _Float16 smem[2 * PW];
  const int lane = threadIdx.x & 31;
  const int wid = threadIdx.x >> 5;
  const int m0 = (blockIdx.x * 2 + wid) * 16;  // batch row base for this wave
  const int n = lane & 15;                     // D-layout column / B column
  const int hi = lane >> 4;
  const int mb = hi * 8;     // D-layout row base
  const int mr = lane & 15;  // A-layout row

  _Float16* s = smem + wid * PW;
  _Float16* aX = s;                    // 16 x L32 (x_t padded to K=32)
  _Float16* aH = aX + 16 * L32;        // h (previous, level1/level4 input)
  _Float16* aHH = aH + 16 * L32;       // hht (level2 input)
  _Float16* aN = aHH + 16 * L32;       // nt
  _Float16* aD = aN + 16 * L64;        // dt
  _Float16* aC0 = aD + 16 * L64;       // Ct double buffer
  _Float16* aC1 = aC0 + 16 * L128;
  _Float16* aS = aC1 + 16 * L128;      // st
  _Float16* aT0 = aS + 16 * L128;      // Ht double buffer
  _Float16* aT1 = aT0 + 16 * L128;

  // ws offsets (halves), gate order f,i,o,c (level4: i,o,c)
  constexpr int OW1[4] = {0, 1024, 2048, 3072};
  constexpr int OU1[4] = {4096, 5120, 6144, 7168};
  constexpr int OW2[4] = {8192, 10240, 12288, 14336};
  constexpr int OU2[4] = {16384, 20480, 24576, 28672};
  constexpr int OW3[4] = {32768, 40960, 49152, 57344};
  constexpr int OU3[4] = {65536, 81920, 98304, 114688};
  constexpr int OVO = 131072;
  constexpr int OW4[3] = {147456, 151552, 155648};
  constexpr int OU4[3] = {159744, 176128, 192512};

  // zero the K-pad columns of aX once
  for (int i = lane; i < 256; i += 32) {
    int m = i >> 4, c = i & 15;
    aX[m * L32 + 16 + c] = (_Float16)0.f;
  }

  // f32 recurrent state held in registers (D layout tiles)
  v8f ct[2], htp[2], nt_[4], st_[8];

#pragma unroll 1
  for (int t = 0; t < T_STEPS; ++t) {
    const bool first = (t == 0);
    const _Float16* aCR = (t & 1) ? aC1 : aC0;
    _Float16* aCW = (t & 1) ? aC0 : aC1;
    const _Float16* aTR = (t & 1) ? aT1 : aT0;
    _Float16* aTW = (t & 1) ? aT0 : aT1;

    // stage x_t tile (16 rows x 16 cols, f32 -> f16)
    for (int i = lane; i < 256; i += 32) {
      int m = i >> 4, c = i & 15;
      aX[m * L32 + c] =
          (_Float16)A.x[((size_t)(m0 + m) * T_STEPS + t) * D_IN + c];
    }

    // Hoisted A fragments: loaded once per timestep, reused by every WMMA.
    v16h Ax = lda_frag(aX, L32, 0, mr, hi);
    v16h Ahp = first ? Ax : lda_frag(aH, L32, 0, mr, hi);  // h_{t-1}

    // ---------------- Level 1 (H1 = 32, 2 N-tiles) ----------------
    auto g1tile = [&](int g, int j) -> v8f {
      v8f acc = vsplat(A.b1[g][j * 16 + n]);
      acc = wmma_(Ax, ldb_frag(A.wb + OW1[g], 32, j * 16 + n, 0, hi), acc);
      if (!first)
        acc = wmma_(Ahp, ldb_frag(A.wb + OU1[g], 32, j * 16 + n, 0, hi), acc);
      return acc;
    };

    v8f ht1[2];
#pragma unroll
    for (int j = 0; j < 2; ++j) {
      v8f fg = vsplat(0.f);
      if (!first) fg = vsig(g1tile(0, j));
      v8f ig = vsig(g1tile(1, j));
      v8f og = vsig(g1tile(2, j));
      v8f gg = vtanh(g1tile(3, j));
      v8f c_ = first ? ig * gg : fg * ct[j] + ig * gg;
      ct[j] = c_;
      ht1[j] = og * vtanh(c_);
    }
#pragma unroll
    for (int j = 0; j < 2; ++j) {
      v8f l2in = first ? ht1[j] : ht1[j] - htp[j];  // hht (or ht at t=0)
      st_tile(aHH, L32, j, n, mb, l2in);
      if (first) st_tile(aH, L32, j, n, mb, ht1[j]);  // level4 uses ht at t=0
      htp[j] = ht1[j];
    }

    v16h Ahh = lda_frag(aHH, L32, 0, mr, hi);
    // aH: at t=0 just written with ht (== level-4 input); t>=1 unchanged == Ahp
    v16h Ah4 = first ? lda_frag(aH, L32, 0, mr, hi) : Ahp;
    v16h An[2];
    An[0] = first ? Ax : lda_frag(aN, L64, 0, mr, hi);  // OLD nt
    An[1] = first ? Ax : lda_frag(aN, L64, 1, mr, hi);

    // ---------------- Level 2 (HN = 64, 4 N-tiles) ----------------
    auto g2tile = [&](int g, int j, bool useN) -> v8f {
      v8f acc = vsplat(A.b2[g][j * 16 + n]);
      acc = wmma_(Ahh, ldb_frag(A.wb + OW2[g], 32, j * 16 + n, 0, hi), acc);
      if (useN) {
#pragma unroll
        for (int kt = 0; kt < 2; ++kt)
          acc = wmma_(An[kt], ldb_frag(A.wb + OU2[g], 64, j * 16 + n, kt, hi),
                      acc);
      }
      return acc;
    };
#pragma unroll
    for (int j = 0; j < 4; ++j) {  // f1,i1,g1 read OLD nt
      v8f f1 = vsplat(0.f);
      if (!first) f1 = vsig(g2tile(0, j, true));
      v8f i1 = vsig(g2tile(1, j, !first));
      v8f g1v = vtanh(g2tile(3, j, !first));
      nt_[j] = first ? i1 * g1v : f1 * nt_[j] + i1 * g1v;
    }
#pragma unroll
    for (int j = 0; j < 4; ++j) st_tile(aN, L64, j, n, mb, nt_[j]);  // NEW nt
    An[0] = lda_frag(aN, L64, 0, mr, hi);  // reload NEW nt
    An[1] = lda_frag(aN, L64, 1, mr, hi);
#pragma unroll
    for (int j = 0; j < 4; ++j) {  // ot1 reads NEW nt (matches reference)
      v8f o1 = vsig(g2tile(2, j, !first));
      v8f dt = o1 * vtanh(nt_[j]);
      st_tile(aD, L64, j, n, mb, dt);
    }

    v16h Ad[2];
    Ad[0] = lda_frag(aD, L64, 0, mr, hi);
    Ad[1] = lda_frag(aD, L64, 1, mr, hi);
    v16h Ac[4];  // OLD Ct (read buffer untouched this step: valid throughout)
#pragma unroll
    for (int kt = 0; kt < 4; ++kt)
      Ac[kt] = first ? Ax : lda_frag(aCR, L128, kt, mr, hi);

    // ---------------- Level 3 (HS = 128, 8 N-tiles) ----------------
    auto g3tile = [&](int g, int j, bool useC) -> v8f {
      v8f acc = vsplat(A.b3[g][j * 16 + n]);
#pragma unroll
      for (int kt = 0; kt < 2; ++kt)
        acc = wmma_(Ad[kt], ldb_frag(A.wb + OW3[g], 64, j * 16 + n, kt, hi),
                    acc);
      if (useC) {
#pragma unroll
        for (int kt = 0; kt < 4; ++kt)
          acc = wmma_(Ac[kt], ldb_frag(A.wb + OU3[g], 128, j * 16 + n, kt, hi),
                      acc);
      }
      return acc;
    };
#pragma unroll
    for (int j = 0; j < 8; ++j) {  // f2,i2,g2 (old Ct)
      v8f f2 = vsplat(0.f);
      if (!first) f2 = vsig(g3tile(0, j, true));
      v8f i2 = vsig(g3tile(1, j, !first));
      v8f g2 = vtanh(g3tile(3, j, !first));
      st_[j] = first ? i2 * g2 : f2 * st_[j] + i2 * g2;
    }
#pragma unroll
    for (int j = 0; j < 8; ++j) st_tile(aS, L128, j, n, mb, st_[j]);  // NEW st

    v16h As[4];
#pragma unroll
    for (int kt = 0; kt < 4; ++kt) As[kt] = lda_frag(aS, L128, kt, mr, hi);
    v16h At[4];  // OLD Ht (read buffer untouched this step)
#pragma unroll
    for (int kt = 0; kt < 4; ++kt)
      At[kt] = first ? Ax : lda_frag(aTR, L128, kt, mr, hi);

    // ---------------- Level 3 ot2 + Level 4 fused ----------------
    auto g4tile = [&](int g, int j, bool useT) -> v8f {
      v8f acc = vsplat(A.b4[g][j * 16 + n]);
      acc = wmma_(Ah4, ldb_frag(A.wb + OW4[g], 32, j * 16 + n, 0, hi), acc);
      if (useT) {
#pragma unroll
        for (int kt = 0; kt < 4; ++kt)
          acc = wmma_(At[kt], ldb_frag(A.wb + OU4[g], 128, j * 16 + n, kt, hi),
                      acc);
      }
      return acc;
    };

#pragma unroll
    for (int j = 0; j < 8; ++j) {
      // ot2 = sig(dt@wo2 [+ Ct@uo2] + st_new@vo + bo2)
      v8f acc = vsplat(A.b3[2][j * 16 + n]);
#pragma unroll
      for (int kt = 0; kt < 2; ++kt)
        acc = wmma_(Ad[kt], ldb_frag(A.wb + OW3[2], 64, j * 16 + n, kt, hi),
                    acc);
      if (!first) {
#pragma unroll
        for (int kt = 0; kt < 4; ++kt)
          acc = wmma_(Ac[kt], ldb_frag(A.wb + OU3[2], 128, j * 16 + n, kt, hi),
                      acc);
      }
#pragma unroll
      for (int kt = 0; kt < 4; ++kt)
        acc = wmma_(As[kt], ldb_frag(A.wb + OVO, 128, j * 16 + n, kt, hi), acc);
      v8f o2 = vsig(acc);
      v8f tt = o2 * vtanh(st_[j]);

      // level 4 (reads OLD h and OLD Ht)
      v8f i3 = vsig(g4tile(0, j, !first));
      v8f g3 = vtanh(g4tile(2, j, !first));
      v8f Cn = tt + i3 * g3;
      v8f o3 = vsig(g4tile(1, j, !first));
      v8f Hn = o3 * vtanh(Cn);

      st_tile(aCW, L128, j, n, mb, Cn);
      st_tile(aTW, L128, j, n, mb, Hn);
#pragma unroll
      for (int r = 0; r < 8; ++r)
        A.out[((size_t)(m0 + mb + r) * T_STEPS + t) * 128 + j * 16 + n] =
            Hn[r];
    }

    // h_t becomes "previous h" for next timestep
#pragma unroll
    for (int j = 0; j < 2; ++j) st_tile(aH, L32, j, n, mb, htp[j]);
  }
}

// ---------------------------------------------------------------------------
// launch
// ---------------------------------------------------------------------------
extern "C" void kernel_launch(void* const* d_in, const int* in_sizes, int n_in,
                              void* d_out, int out_size, void* d_ws,
                              size_t ws_size, hipStream_t stream) {
  (void)n_in;
  (void)out_size;
  (void)ws_size;

  // params dict order (after x at d_in[0]):
  // per g in {f,i,o,c}: w,u,b, w1,u1,b1, w2,u2,b2  -> 9 entries each
  // then vo; then per g in {i,o,c}: w3,u3,b3
  PrepArgs P;
  int id = 0;
  auto W = [&](int din, int K, int N, int off) {
    P.src[id] = (const float*)d_in[din];
    P.K[id] = K;
    P.N[id] = N;
    P.off[id] = off;
    ++id;
  };
  const int w1i[4] = {1, 10, 19, 28};
  const int u1i[4] = {2, 11, 20, 29};
  const int w2i[4] = {4, 13, 22, 31};
  const int u2i[4] = {5, 14, 23, 32};
  const int w3i[4] = {7, 16, 25, 34};
  const int u3i[4] = {8, 17, 26, 35};
  const int w4i[3] = {38, 41, 44};
  const int u4i[3] = {39, 42, 45};
  for (int g = 0; g < 4; ++g) W(w1i[g], 16, 32, 0 + 1024 * g);
  for (int g = 0; g < 4; ++g) W(u1i[g], 32, 32, 4096 + 1024 * g);
  for (int g = 0; g < 4; ++g) W(w2i[g], 32, 64, 8192 + 2048 * g);
  for (int g = 0; g < 4; ++g) W(u2i[g], 64, 64, 16384 + 4096 * g);
  for (int g = 0; g < 4; ++g) W(w3i[g], 64, 128, 32768 + 8192 * g);
  for (int g = 0; g < 4; ++g) W(u3i[g], 128, 128, 65536 + 16384 * g);
  W(37, 128, 128, 131072);  // vo
  for (int g = 0; g < 3; ++g) W(w4i[g], 32, 128, 147456 + 4096 * g);
  for (int g = 0; g < 3; ++g) W(u4i[g], 128, 128, 159744 + 16384 * g);

  _Float16* wb = (_Float16*)d_ws;
  MIM_prep_kernel<<<NMAT, 256, 0, stream>>>(P, wb);

  MainArgs M;
  M.x = (const float*)d_in[0];
  M.out = (float*)d_out;
  M.wb = wb;
  const int b1i[4] = {3, 12, 21, 30};
  const int b2i[4] = {6, 15, 24, 33};
  const int b3i[4] = {9, 18, 27, 36};
  const int b4i[3] = {40, 43, 46};
  for (int g = 0; g < 4; ++g) {
    M.b1[g] = (const float*)d_in[b1i[g]];
    M.b2[g] = (const float*)d_in[b2i[g]];
    M.b3[g] = (const float*)d_in[b3i[g]];
  }
  for (int g = 0; g < 3; ++g) M.b4[g] = (const float*)d_in[b4i[g]];

  int B = in_sizes[0] / (T_STEPS * D_IN);
  int nblk = B / 32;  // 2 waves/block * 16 batch rows/wave
  MIM_lstm_kernel<<<nblk, 64, 0, stream>>>(M);
}
